// NewsEncoder_81303730913793
// MI455X (gfx1250) — matrix operand order repaired and checked by reading, not compiled, not run on hardware
//
#include <hip/hip_runtime.h>
#include <hip/hip_bf16.h>

// Problem constants (match reference)
#define SEQ   30
#define SPAD  32      // padded sequence rows (rows 30,31 pre-zeroed in Ebf)
#define DM    300
#define DPAD  320     // K padded to 10 x 32
#define NQPAD 304     // q N padded to 19 x 16
#define NT_Q  19
#define KTN   10
#define NH    20
#define HD    20
#define ATT   200
#define BATCH 4096
#define HDIM  400     // NH*HD

typedef __attribute__((ext_vector_type(16))) __bf16 v16bf;
typedef __attribute__((ext_vector_type(8)))  float  v8f;

static __device__ inline v8f v8zero() {
  v8f z;
#pragma unroll
  for (int i = 0; i < 8; ++i) z[i] = 0.0f;
  return z;
}

static __device__ inline v8f wmma_bf16(v16bf a, v16bf b, v8f c) {
  // D = A x B + C, 16x16x32 bf16 -> f32
  return __builtin_amdgcn_wmma_f32_16x16x32_bf16(false, a, false, b, (short)0, c, false, false);
}

// A fragment (this code's contiguous k order, shared with the B loaders/packs): lane L holds row M=m0+(L&15);
// lanes 0-15: K = {0..15}, lanes 16-31: K = {16..31} (relative k0).
static __device__ inline v16bf load_a(const __bf16* base, int ld, int m0, int k0, int lane) {
  const __bf16* p = base + (m0 + (lane & 15)) * ld + k0 + ((lane < 16) ? 0 : 16);
  v16bf a;
#pragma unroll
  for (int j = 0; j < 16; ++j) a[j] = p[j];
  return a;
}

// B fragment where B[k][n] = base[n*ld + k]  (i.e. B = M^T of a row-major M).
static __device__ inline v16bf load_bt(const __bf16* base, int ld, int k0, int n0, int lane) {
  const __bf16* p = base + (n0 + (lane & 15)) * ld + k0 + ((lane < 16) ? 0 : 16);
  v16bf v;
#pragma unroll
  for (int j = 0; j < 16; ++j) v[j] = p[j];
  return v;
}

// B fragment where B[k][n] = base[k*ld + n] (row-major B).
static __device__ inline v16bf load_brm(const __bf16* base, int ld, int k0, int n0, int lane) {
  const __bf16* p = base + (k0 + ((lane < 16) ? 0 : 16)) * ld + (n0 + (lane & 15));
  v16bf v;
#pragma unroll
  for (int j = 0; j < 16; ++j) v[j] = p[j * ld];
  return v;
}

// ---------------- prep kernels ----------------

// Ebf[b][s][0..DPAD) = bf16(emb[x[b,s]] + pe[s]); rows s in [SEQ,SPAD) and cols >= DM zeroed.
__global__ void prep_e_kernel(const int* __restrict__ x, const float* __restrict__ emb,
                              const float* __restrict__ pe, __bf16* __restrict__ Ebf) {
  int row = blockIdx.x;                 // 0 .. BATCH*SPAD-1
  int s = row & (SPAD - 1);
  int b = row >> 5;
  __bf16* dst = Ebf + (long)row * DPAD;
  if (s >= SEQ) {
    for (int d = threadIdx.x; d < DPAD; d += blockDim.x) dst[d] = (__bf16)0.0f;
    return;
  }
  long idx = (long)x[b * SEQ + s];
  const float* er = emb + idx * DM;
  const float* pr = pe + s * DM;
  for (int d = threadIdx.x; d < DPAD; d += blockDim.x) {
    float v = (d < DM) ? (er[d] + pr[d]) : 0.0f;
    dst[d] = (__bf16)v;
  }
}

// Pack Wq into WMMA B-fragment order: B[k=d][n=o] = Wq[h][o][d]
// out[(((h*KTN+kt)*NT_Q+nt)*32+lane)*16 + j], k = kt*32 + (lane<16?0:16) + j, n = nt*16 + (lane&15)
__global__ void prep_wq_kernel(const float* __restrict__ Wq, __bf16* __restrict__ out) {
  int id = blockIdx.x * blockDim.x + threadIdx.x;   // 20*10*19*512 = 1,945,600
  int j = id & 15;
  int lane = (id >> 4) & 31;
  int nt = (id >> 9) % NT_Q;
  int kt = (id / (512 * NT_Q)) % KTN;
  int h  = id / (512 * NT_Q * KTN);
  if (h >= NH) return;
  int k = kt * 32 + ((lane < 16) ? 0 : 16) + j;
  int n = nt * 16 + (lane & 15);
  float v = (k < DM && n < DM) ? Wq[((long)h * DM + n) * DM + k] : 0.0f;
  out[id] = (__bf16)v;
}

// Pack Wv similarly: B[k=d][n=o] = Wv[h][o][d], o padded 20->32 (2 n-tiles)
__global__ void prep_wv_kernel(const float* __restrict__ Wv, __bf16* __restrict__ out) {
  int id = blockIdx.x * blockDim.x + threadIdx.x;   // 20*10*2*512 = 204,800
  int j = id & 15;
  int lane = (id >> 4) & 31;
  int nt = (id >> 9) & 1;
  int kt = (id >> 10) % KTN;
  int h  = id / (1024 * KTN);
  if (h >= NH) return;
  int k = kt * 32 + ((lane < 16) ? 0 : 16) + j;
  int n = nt * 16 + (lane & 15);
  float v = (k < DM && n < HD) ? Wv[((long)h * HD + n) * DM + k] : 0.0f;
  out[id] = (__bf16)v;
}

// ---------------- fused attention kernel (one wave per (batch, head)) ----------------

__global__ void __launch_bounds__(32)
attn_kernel(const __bf16* __restrict__ Ebf, const __bf16* __restrict__ wqpack,
            const __bf16* __restrict__ wvpack, const float* __restrict__ bq,
            const float* __restrict__ bv, float* __restrict__ hbuf) {
  __shared__ alignas(16) __bf16 e_lds[SPAD * DPAD];    // 20 KB
  __shared__ alignas(16) __bf16 q_lds[SPAD * DPAD];    // 20 KB
  __shared__ alignas(16) __bf16 attn_lds[32 * 32];     // 2 KB
  __shared__ alignas(16) __bf16 ev_lds[32 * 32];       // 2 KB

  const int b = blockIdx.x;
  const int h = blockIdx.y;
  const int lane = threadIdx.x;
  const int moff = (lane < 16) ? 0 : 8;
  const int nA = lane & 15;
  const int nB = 16 + nA;

  // ---- stage e via async global->LDS DMA path (32 lanes x 16B = 512B per issue)
  {
    unsigned lds_base = (unsigned)(size_t)(&e_lds[0]);
    unsigned long long gbase = (unsigned long long)(Ebf + (long)b * SPAD * DPAD);
    unsigned off = lane * 16u;
#pragma unroll 4
    for (int it = 0; it < (SPAD * DPAD * 2) / 512; ++it) {
      asm volatile("global_load_async_to_lds_b128 %0, %1, %2"
                   :: "v"(lds_base + off), "v"(off), "s"(gbase)
                   : "memory");
      off += 512u;
    }
  }
  // zero q pad columns [NQPAD, DPAD) while the async copy is in flight
  for (int i = lane; i < SPAD * (DPAD - NQPAD); i += 32) {
    int r = i / (DPAD - NQPAD);
    int c = NQPAD + i % (DPAD - NQPAD);
    q_lds[r * DPAD + c] = (__bf16)0.0f;
  }
  asm volatile("s_wait_asynccnt 0x0" ::: "memory");
  __syncthreads();

  // ---- q = e @ Wq[h]^T + bq   (M=32, N=304, K=320) -> LDS bf16
  const v16bf* wq = (const v16bf*)wqpack + (long)h * KTN * NT_Q * 32;
  for (int nt = 0; nt < NT_Q; ++nt) {
    v8f acc0 = v8zero(), acc1 = v8zero();
    for (int kt = 0; kt < KTN; ++kt) {
      v16bf bfrag = wq[(kt * NT_Q + nt) * 32 + lane];
      v16bf a0 = load_a(e_lds, DPAD, 0, kt * 32, lane);
      v16bf a1 = load_a(e_lds, DPAD, 16, kt * 32, lane);
      acc0 = wmma_bf16(a0, bfrag, acc0);
      acc1 = wmma_bf16(a1, bfrag, acc1);
    }
    int n = nt * 16 + nA;
    float bias = (n < DM) ? bq[h * DM + n] : 0.0f;
#pragma unroll
    for (int r = 0; r < 8; ++r) {
      float v0 = (n < DM) ? acc0[r] + bias : 0.0f;
      float v1 = (n < DM) ? acc1[r] + bias : 0.0f;
      q_lds[(r + moff) * DPAD + n] = (__bf16)v0;
      q_lds[(16 + r + moff) * DPAD + n] = (__bf16)v1;
    }
  }
  __syncthreads();

  // ---- scores = q @ e^T  (M=32 query s, N=32 key t, K=320)
  v8f s00 = v8zero(), s01 = v8zero(), s10 = v8zero(), s11 = v8zero();
  for (int kt = 0; kt < KTN; ++kt) {
    v16bf a0 = load_a(q_lds, DPAD, 0, kt * 32, lane);
    v16bf a1 = load_a(q_lds, DPAD, 16, kt * 32, lane);
    v16bf b0 = load_bt(e_lds, DPAD, kt * 32, 0, lane);
    v16bf b1 = load_bt(e_lds, DPAD, kt * 32, 16, lane);
    s00 = wmma_bf16(a0, b0, s00);
    s01 = wmma_bf16(a0, b1, s01);
    s10 = wmma_bf16(a1, b0, s10);
    s11 = wmma_bf16(a1, b1, s11);
  }

  // ---- softmax over the query axis (a column of the C layout spans lanes L and L^16)
  float mA = -3.0e38f, mB = -3.0e38f;
#pragma unroll
  for (int r = 0; r < 8; ++r) {
    mA = fmaxf(mA, s00[r]);
    mB = fmaxf(mB, s01[r]);
    if (16 + r + moff < SEQ) { mA = fmaxf(mA, s10[r]); mB = fmaxf(mB, s11[r]); }
  }
  mA = fmaxf(mA, __shfl_xor(mA, 16, 32));
  mB = fmaxf(mB, __shfl_xor(mB, 16, 32));
  float p00[8], p01[8], p10[8], p11[8];
  float sumA = 0.0f, sumB = 0.0f;
#pragma unroll
  for (int r = 0; r < 8; ++r) {
    p00[r] = __expf(s00[r] - mA); sumA += p00[r];
    p01[r] = __expf(s01[r] - mB); sumB += p01[r];
    bool ok = (16 + r + moff) < SEQ;
    p10[r] = ok ? __expf(s10[r] - mA) : 0.0f; sumA += p10[r];
    p11[r] = ok ? __expf(s11[r] - mB) : 0.0f; sumB += p11[r];
  }
  sumA += __shfl_xor(sumA, 16, 32);
  sumB += __shfl_xor(sumB, 16, 32);
  float iA = 1.0f / sumA, iB = 1.0f / sumB;
  bool tBok = nB < SEQ;   // zero key columns t >= 30 so hv ignores them
#pragma unroll
  for (int r = 0; r < 8; ++r) {
    attn_lds[(r + moff) * 32 + nA] = (__bf16)(p00[r] * iA);
    attn_lds[(16 + r + moff) * 32 + nA] = (__bf16)(p10[r] * iA);
    attn_lds[(r + moff) * 32 + nB] = tBok ? (__bf16)(p01[r] * iB) : (__bf16)0.0f;
    attn_lds[(16 + r + moff) * 32 + nB] = tBok ? (__bf16)(p11[r] * iB) : (__bf16)0.0f;
  }

  // ---- ev = e @ Wv[h]^T   (M=32 t, N=32 o, K=320)   [hv = attn @ ev == (attn@e)@Wv^T]
  const v16bf* wv = (const v16bf*)wvpack + (long)h * KTN * 2 * 32;
  v8f v00 = v8zero(), v01 = v8zero(), v10 = v8zero(), v11 = v8zero();
  for (int kt = 0; kt < KTN; ++kt) {
    v16bf a0 = load_a(e_lds, DPAD, 0, kt * 32, lane);
    v16bf a1 = load_a(e_lds, DPAD, 16, kt * 32, lane);
    v16bf b0 = wv[(kt * 2 + 0) * 32 + lane];
    v16bf b1 = wv[(kt * 2 + 1) * 32 + lane];
    v00 = wmma_bf16(a0, b0, v00);
    v01 = wmma_bf16(a0, b1, v01);
    v10 = wmma_bf16(a1, b0, v10);
    v11 = wmma_bf16(a1, b1, v11);
  }
#pragma unroll
  for (int r = 0; r < 8; ++r) {
    ev_lds[(r + moff) * 32 + nA] = (__bf16)v00[r];
    ev_lds[(16 + r + moff) * 32 + nA] = (__bf16)v10[r];
    ev_lds[(r + moff) * 32 + nB] = (__bf16)v01[r];
    ev_lds[(16 + r + moff) * 32 + nB] = (__bf16)v11[r];
  }
  __syncthreads();

  // ---- hv = attn @ ev + bv   (M=32 s, N=32 o, K=32 t)
  {
    v16bf a0 = load_a(attn_lds, 32, 0, 0, lane);
    v16bf a1 = load_a(attn_lds, 32, 16, 0, lane);
    v16bf b0 = load_brm(ev_lds, 32, 0, 0, lane);
    v16bf b1 = load_brm(ev_lds, 32, 0, 16, lane);
    v8f h00 = wmma_bf16(a0, b0, v8zero());
    v8f h01 = wmma_bf16(a0, b1, v8zero());
    v8f h10 = wmma_bf16(a1, b0, v8zero());
    v8f h11 = wmma_bf16(a1, b1, v8zero());

    float bvA = (nA < HD) ? bv[h * HD + nA] : 0.0f;
    float bvB = (nB < HD) ? bv[h * HD + nB] : 0.0f;
    float* hrow = hbuf + (long)b * SEQ * HDIM + h * HD;
#pragma unroll
    for (int r = 0; r < 8; ++r) {
      int s0 = r + moff;
      int s1 = 16 + r + moff;
      if (nA < HD) {
        if (s0 < SEQ) hrow[s0 * HDIM + nA] = h00[r] + bvA;
        if (s1 < SEQ) hrow[s1 * HDIM + nA] = h10[r] + bvA;
      }
      if (nB < HD) {
        if (s0 < SEQ) hrow[s0 * HDIM + nB] = h01[r] + bvB;
        if (s1 < SEQ) hrow[s1 * HDIM + nB] = h11[r] + bvB;
      }
    }
  }
}

// ---------------- additive-attention tail (VALU; deterministic, no atomics) ----------------

__global__ void __launch_bounds__(256)
tail_kernel(const float* __restrict__ hbuf, const float* __restrict__ Wa,
            const float* __restrict__ ba, const float* __restrict__ wq2,
            const float* __restrict__ bq2, float* __restrict__ out) {
  __shared__ alignas(16) float hs[SEQ * HDIM];   // 48 KB
  __shared__ float part[SEQ * 8];
  __shared__ float a_lds[SEQ];
  int b = blockIdx.x, tid = threadIdx.x;
  const float4* src = (const float4*)(hbuf + (long)b * SEQ * HDIM);
  float4* dstv = (float4*)hs;
  for (int i = tid; i < (SEQ * HDIM) / 4; i += 256) dstv[i] = src[i];
  __syncthreads();

  if (tid < SEQ * 8) {
    int s = tid >> 3, g = tid & 7;
    const float* hr = hs + s * HDIM;
    float acc = 0.0f;
    for (int u = g; u < ATT; u += 8) {
      const float* wr = Wa + u * HDIM;
      float d = ba[u];
      for (int j = 0; j < HDIM; ++j) d += hr[j] * wr[j];
      acc += wq2[u] * tanhf(d);
    }
    part[tid] = acc;
  }
  __syncthreads();
  if (tid < SEQ) {
    float a = bq2[0];
#pragma unroll
    for (int g = 0; g < 8; ++g) a += part[tid * 8 + g];
    a_lds[tid] = a;
  }
  __syncthreads();
  for (int d = tid; d < HDIM; d += 256) {
    float z = 0.0f;
#pragma unroll 5
    for (int s = 0; s < SEQ; ++s) z += a_lds[s] * hs[s * HDIM + d];
    out[(long)b * HDIM + d] = z;
  }
}

// ---------------- launcher ----------------

extern "C" void kernel_launch(void* const* d_in, const int* in_sizes, int n_in,
                              void* d_out, int out_size, void* d_ws, size_t ws_size,
                              hipStream_t stream) {
  const int*   x   = (const int*)d_in[0];
  const float* emb = (const float*)d_in[1];
  const float* pe  = (const float*)d_in[2];
  const float* Wq  = (const float*)d_in[3];
  const float* bq  = (const float*)d_in[4];
  const float* Wv  = (const float*)d_in[5];
  const float* bv  = (const float*)d_in[6];
  const float* Wa  = (const float*)d_in[7];
  const float* ba  = (const float*)d_in[8];
  const float* wq2 = (const float*)d_in[9];
  const float* bq2 = (const float*)d_in[10];
  (void)in_sizes; (void)n_in; (void)out_size; (void)ws_size;

  char* ws = (char*)d_ws;
  __bf16* Ebf = (__bf16*)ws;                          // 4096*32*320*2   = 83,886,080 B
  __bf16* Wqp = (__bf16*)(ws + 83886080);             // 20*10*19*512*2  =  3,891,200 B
  __bf16* Wvp = (__bf16*)(ws + 87777280);             // 20*10*2*512*2   =    409,600 B
  float*  hbuf = (float*)(ws + 88186880);             // 4096*30*400*4   = 196,608,000 B

  prep_e_kernel<<<BATCH * SPAD, 64, 0, stream>>>(x, emb, pe, Ebf);
  prep_wq_kernel<<<1945600 / 256, 256, 0, stream>>>(Wq, Wqp);
  prep_wv_kernel<<<204800 / 256, 256, 0, stream>>>(Wv, Wvp);

  dim3 g(BATCH, NH);
  attn_kernel<<<g, 32, 0, stream>>>(Ebf, Wqp, Wvp, bq, bv, hbuf);

  tail_kernel<<<BATCH, 256, 0, stream>>>(hbuf, Wa, ba, wq2, bq2, (float*)d_out);
}
